// PointNet2SSGSeg_32263794328109
// MI455X (gfx1250) — matrix operand order repaired and checked by reading, not compiled, not run on hardware
//
#include <hip/hip_runtime.h>
#include <stdint.h>

// PointNet++ SSG seg forward for MI455X (gfx1250, wave32).
// Conv MLPs mapped to v_wmma_f32_16x16x32_bf16 (bf16 operands, f32 accum):
//   A = W row-major [Opad][Cpad] bf16, async-copied (global->LDS, ASYNCcnt) per block,
//   B = X point-major [L][Cpad] bf16 from global, double-buffered in VGPRs.
// Each wave computes a 32x32 output block (2x2 WMMA tiles).

#define B_SZ 16
#define N0   4096
#define EPS_BN 1e-5f

typedef __attribute__((ext_vector_type(16))) __bf16 bf16x16;
typedef __attribute__((ext_vector_type(8)))  float  f32x8;

union Frag { uint4 q[2]; bf16x16 v; };

__device__ __forceinline__ uint16_t f2bf(float f) {
  union { float f; uint32_t u; } c; c.f = f;
  uint32_t u = c.u;
  u += 0x7FFFu + ((u >> 16) & 1u);   // round-to-nearest-even
  return (uint16_t)(u >> 16);
}

__device__ __forceinline__ f32x8 wmma_bf16(const Frag& a, const Frag& b, f32x8 c) {
  return __builtin_amdgcn_wmma_f32_16x16x32_bf16(
      /*neg_a=*/false, a.v, /*neg_b=*/false, b.v,
      /*c_mod=*/(short)0, c, /*reuse_a=*/false, /*reuse_b=*/false);
}

// B fragment (32x16 bf16): lane holds col N; regs 0-3: K = 8*hi..+7; regs 4-7: K = 16+8*hi..+7
__device__ __forceinline__ void loadB(Frag& b0, Frag& b1,
                                      const uint16_t* xr0, const uint16_t* xr1, int k) {
  b0.q[0] = *(const uint4*)(xr0 + k*32);
  b0.q[1] = *(const uint4*)(xr0 + k*32 + 16);
  b1.q[0] = *(const uint4*)(xr1 + k*32);
  b1.q[1] = *(const uint4*)(xr1 + k*32 + 16);
}

// A fragment (16x32 bf16): lane holds row M; regs 0-3: K = 8*hi..+7; regs 4-7: K = 16+8*hi..+7
__device__ __forceinline__ void loadA(Frag& a0, Frag& a1,
                                      const uint16_t* s0, const uint16_t* s1, int k) {
  a0.q[0] = *(const uint4*)(s0 + k*32);
  a0.q[1] = *(const uint4*)(s0 + k*32 + 16);
  a1.q[0] = *(const uint4*)(s1 + k*32);
  a1.q[1] = *(const uint4*)(s1 + k*32 + 16);
}

// ---------------- extract xyz / feat from pointcloud (B,N,6) ----------------
__global__ void kextract(const float* __restrict__ pc, float* __restrict__ xyz,
                         float* __restrict__ feat, int total) {
  int i = blockIdx.x * blockDim.x + threadIdx.x;
  if (i >= total) return;
  const float* p = pc + (size_t)i * 6;
  xyz[i*3+0] = p[0]; xyz[i*3+1] = p[1]; xyz[i*3+2] = p[2];
  feat[i*3+0] = p[3]; feat[i*3+1] = p[4]; feat[i*3+2] = p[5];
}

// ---------------- farthest point sampling: one block per batch ----------------
__global__ void kfps(const float* __restrict__ xyz, int Nin, int M, int* __restrict__ out) {
  __shared__ float dist[4096];
  __shared__ float rmax[256];
  __shared__ int   rarg[256];
  __shared__ int   farSh;
  int b = blockIdx.x, t = threadIdx.x, T = blockDim.x;
  const float* X = xyz + (size_t)b * Nin * 3;
  for (int n = t; n < Nin; n += T) dist[n] = 1e10f;
  if (t == 0) farSh = 0;
  __syncthreads();
  for (int j = 0; j < M; j++) {
    int far = farSh;
    if (t == 0) out[b*M + j] = far;
    float cx = X[far*3], cy = X[far*3+1], cz = X[far*3+2];
    float lb = -1.f; int la = 0;
    for (int n = t; n < Nin; n += T) {
      float dx = X[n*3]-cx, dy = X[n*3+1]-cy, dz = X[n*3+2]-cz;
      float d = dx*dx + dy*dy + dz*dz;
      float dm = fminf(dist[n], d);
      dist[n] = dm;
      if (dm > lb) { lb = dm; la = n; }
    }
    rmax[t] = lb; rarg[t] = la;
    __syncthreads();
    for (int s = T >> 1; s > 0; s >>= 1) {
      if (t < s && rmax[t+s] > rmax[t]) { rmax[t] = rmax[t+s]; rarg[t] = rarg[t+s]; }
      __syncthreads();
    }
    if (t == 0) farSh = rarg[0];
    __syncthreads();
  }
}

__global__ void kgather_xyz(const float* __restrict__ xyz_in, int Nin,
                            const int* __restrict__ idx, int M, float* __restrict__ xyz_out) {
  int i = blockIdx.x * blockDim.x + threadIdx.x;
  if (i >= B_SZ * M) return;
  int b = i / M;
  int src = idx[i];
  const float* p = xyz_in + ((size_t)b * Nin + src) * 3;
  xyz_out[i*3] = p[0]; xyz_out[i*3+1] = p[1]; xyz_out[i*3+2] = p[2];
}

// ---------------- 32-NN per query (insertion selection) ----------------
__global__ void kknn(const float* __restrict__ qxyz, int M,
                     const float* __restrict__ sxyz, int Nin, int* __restrict__ nidx) {
  int i = blockIdx.x * blockDim.x + threadIdx.x;
  if (i >= B_SZ * M) return;
  int b = i / M;
  float qx = qxyz[i*3], qy = qxyz[i*3+1], qz = qxyz[i*3+2];
  const float* X = sxyz + (size_t)b * Nin * 3;
  float bd[32]; int bi[32];
  #pragma unroll
  for (int k = 0; k < 32; k++) { bd[k] = 1e30f; bi[k] = 0; }
  for (int n = 0; n < Nin; n++) {
    float dx = X[n*3]-qx, dy = X[n*3+1]-qy, dz = X[n*3+2]-qz;
    float d = dx*dx + dy*dy + dz*dz;
    if (d < bd[31]) {
      int k = 31;
      while (k > 0 && bd[k-1] > d) { bd[k] = bd[k-1]; bi[k] = bi[k-1]; k--; }
      bd[k] = d; bi[k] = n;
    }
  }
  int* o = nidx + (size_t)i * 32;
  for (int k = 0; k < 32; k++) o[k] = bi[k];
}

// ---------------- group: build bf16 X [Lg][Cpad] = [rel-xyz ; gathered feats ; 0pad] ----------------
__global__ void kgroup(const float* __restrict__ sxyz, int Nin,
                       const float* __restrict__ qxyz, int M,
                       const int* __restrict__ nidx,
                       const float* __restrict__ featIn, int Cin,
                       uint16_t* __restrict__ Xbf, int Cpad) {
  int l = blockIdx.x * blockDim.x + threadIdx.x;
  int total = B_SZ * M * 32;
  if (l >= total) return;
  int s = l & 31;
  int q = l >> 5;            // b*M+m
  int b = q / M;
  int src = nidx[(size_t)q * 32 + s];
  const float* sp = sxyz + ((size_t)b * Nin + src) * 3;
  const float* qp = qxyz + (size_t)q * 3;
  uint16_t* o = Xbf + (size_t)l * Cpad;
  o[0] = f2bf(sp[0]-qp[0]); o[1] = f2bf(sp[1]-qp[1]); o[2] = f2bf(sp[2]-qp[2]);
  const float* fp = featIn + ((size_t)b * Nin + src) * Cin;
  for (int c = 0; c < Cin; c++)       o[3+c] = f2bf(fp[c]);
  for (int c = 3+Cin; c < Cpad; c++)  o[c]   = 0;
}

// ---------------- weight convert f32 [O][C] -> bf16 padded [Opad][Cpad] ----------------
__global__ void kconvw(const float* __restrict__ W, int O, int C,
                       uint16_t* __restrict__ Wbf, int Opad, int Cpad) {
  int i = blockIdx.x * blockDim.x + threadIdx.x;
  if (i >= Opad * Cpad) return;
  int o = i / Cpad, c = i % Cpad;
  float v = (o < O && c < C) ? W[(size_t)o * C + c] : 0.f;
  Wbf[i] = f2bf(v);
}

// ---------------- WMMA GEMM: Y[Opad][L] = Wbf[Opad][Cpad] * Xbf[L][Cpad]^T + bias ----------------
// Block = 8 waves, all on the same 32-row W slab, async-copied global->LDS (ASYNCcnt).
// Wave = 32x32 output (2x2 WMMA tiles), B fragments double-buffered.
__global__ void __launch_bounds__(256)
kgemm(const uint16_t* __restrict__ Wbf, const uint16_t* __restrict__ Xbf,
      const float* __restrict__ bias, int O,
      float* __restrict__ Y, int Opad, int Cpad, int L) {
  __shared__ __align__(16) uint16_t sW[32 * 768];   // 48 KB max (Cpad <= 768)
  int nL2 = L >> 5;                 // number of 32-wide L pairs
  int lblocks = (nL2 + 7) >> 3;
  int bo = blockIdx.x / lblocks;
  int lb = blockIdx.x % lblocks;
  int obase = bo * 32;

  // Async-stage the W slab [obase, obase+32) x Cpad into LDS:
  // global_load_async_to_lds_b128 moves 16B per lane cache->LDS, tracked by ASYNCcnt
  // (no VGPR round-trip). Generic shared-pointer low 32 bits = LDS byte address.
  {
    const char* g = (const char*)(Wbf + (size_t)obase * Cpad);
    uint32_t lbase32 = (uint32_t)(uintptr_t)(void*)sW;
    int n16 = (32 * Cpad * 2) >> 4;                // 16-byte chunks
    for (int i = threadIdx.x; i < n16; i += blockDim.x) {
      uint32_t loff = lbase32 + (uint32_t)i * 16;
      uint64_t ga = (uint64_t)(uintptr_t)(g + (size_t)i * 16);
      asm volatile("global_load_async_to_lds_b128 %0, %1, off"
                   :: "v"(loff), "v"(ga) : "memory");
    }
    asm volatile("s_wait_asynccnt 0x0" ::: "memory");
  }
  __syncthreads();

  int wv   = threadIdx.x >> 5;      // wave id in block (wave32)
  int lane = threadIdx.x & 31;
  int lpair = lb * 8 + wv;
  if (lpair >= nL2) return;         // wave-uniform: EXEC all-ones inside
  int lm = lane & 15, hi = lane >> 4;
  int lbase = lpair * 32;

  const uint16_t* xr0 = Xbf + (size_t)(lbase + lm) * Cpad + hi * 8;
  const uint16_t* xr1 = xr0 + (size_t)16 * Cpad;
  const uint16_t* s0  = sW + lm * Cpad + hi * 8;          // A tile rows 0..15
  const uint16_t* s1  = sW + (16 + lm) * Cpad + hi * 8;   // A tile rows 16..31

  f32x8 z = {0.f,0.f,0.f,0.f,0.f,0.f,0.f,0.f};
  f32x8 a00 = z, a01 = z, a10 = z, a11 = z;
  Frag A0, A1, Pa0, Pa1, Pb0, Pb1;   // A frags + ping/pong B frags

  int ks = Cpad >> 5;
  loadB(Pa0, Pa1, xr0, xr1, 0);
  int k = 0;
  for (; k + 1 < ks; k += 2) {
    loadB(Pb0, Pb1, xr0, xr1, k + 1);            // in flight during first 4 WMMAs
    loadA(A0, A1, s0, s1, k);
    a00 = wmma_bf16(A0, Pa0, a00);
    a01 = wmma_bf16(A0, Pa1, a01);
    a10 = wmma_bf16(A1, Pa0, a10);
    a11 = wmma_bf16(A1, Pa1, a11);
    if (k + 2 < ks) {
      loadB(Pa0, Pa1, xr0, xr1, k + 2);          // refill ping during second 4 WMMAs
      __builtin_prefetch((const void*)(xr0 + (k + 3) * 32), 0, 3);
      __builtin_prefetch((const void*)(xr1 + (k + 3) * 32), 0, 3);
    }
    loadA(A0, A1, s0, s1, k + 1);
    a00 = wmma_bf16(A0, Pb0, a00);
    a01 = wmma_bf16(A0, Pb1, a01);
    a10 = wmma_bf16(A1, Pb0, a10);
    a11 = wmma_bf16(A1, Pb1, a11);
  }
  if (k < ks) {
    loadA(A0, A1, s0, s1, k);
    a00 = wmma_bf16(A0, Pa0, a00);
    a01 = wmma_bf16(A0, Pa1, a01);
    a10 = wmma_bf16(A1, Pa0, a10);
    a11 = wmma_bf16(A1, Pa1, a11);
  }

  // D layout: VGPR r, lanes 0-15 -> M=r, lanes 16-31 -> M=r+8; N = lane&15
  #pragma unroll
  for (int r = 0; r < 8; r++) {
    int o0 = obase + r + hi * 8;
    int o1 = o0 + 16;
    float b0 = (o0 < O) ? bias[o0] : 0.f;
    float b1 = (o1 < O) ? bias[o1] : 0.f;
    Y[(size_t)o0 * L + lbase + lm]      = a00[r] + b0;
    Y[(size_t)o0 * L + lbase + 16 + lm] = a01[r] + b0;
    Y[(size_t)o1 * L + lbase + lm]      = a10[r] + b1;
    Y[(size_t)o1 * L + lbase + 16 + lm] = a11[r] + b1;
  }
}

// ---------------- BN statistics: one block per output channel ----------------
__global__ void kbnstats(const float* __restrict__ Y, int L,
                         float* __restrict__ mean, float* __restrict__ rstd) {
  __shared__ float s1[256], s2[256];
  int o = blockIdx.x;
  const float* row = Y + (size_t)o * L;
  float a = 0.f, b = 0.f;
  for (int l = threadIdx.x; l < L; l += blockDim.x) { float v = row[l]; a += v; b += v*v; }
  s1[threadIdx.x] = a; s2[threadIdx.x] = b;
  __syncthreads();
  for (int s = blockDim.x >> 1; s > 0; s >>= 1) {
    if (threadIdx.x < s) { s1[threadIdx.x] += s1[threadIdx.x+s]; s2[threadIdx.x] += s2[threadIdx.x+s]; }
    __syncthreads();
  }
  if (threadIdx.x == 0) {
    float m = s1[0] / (float)L;
    float v = s2[0] / (float)L - m*m;
    mean[o] = m;
    rstd[o] = rsqrtf(fmaxf(v, 0.f) + EPS_BN);
  }
}

// ---------------- BN+ReLU epilogues ----------------
__global__ void kbnrelu_xbf(const float* __restrict__ Y, int O, int L,
                            const float* __restrict__ mean, const float* __restrict__ rstd,
                            uint16_t* __restrict__ Xbf, int Cpad) {
  size_t i = (size_t)blockIdx.x * blockDim.x + threadIdx.x;
  if (i >= (size_t)Cpad * L) return;
  int o = (int)(i / L); int l = (int)(i % L);
  float v = 0.f;
  if (o < O) v = fmaxf((Y[(size_t)o*L + l] - mean[o]) * rstd[o], 0.f);
  Xbf[(size_t)l * Cpad + o] = f2bf(v);
}

__global__ void kbnrelu_y(float* __restrict__ Y, int O, int L,
                          const float* __restrict__ mean, const float* __restrict__ rstd) {
  size_t i = (size_t)blockIdx.x * blockDim.x + threadIdx.x;
  if (i >= (size_t)O * L) return;
  int o = (int)(i / L);
  Y[i] = fmaxf((Y[i] - mean[o]) * rstd[o], 0.f);
}

__global__ void kbnrelu_feat(const float* __restrict__ Y, int O, int L,
                             const float* __restrict__ mean, const float* __restrict__ rstd,
                             float* __restrict__ feat) {
  size_t i = (size_t)blockIdx.x * blockDim.x + threadIdx.x;
  if (i >= (size_t)O * L) return;
  int o = (int)(i / L); int l = (int)(i % L);
  feat[(size_t)l * O + o] = fmaxf((Y[i] - mean[o]) * rstd[o], 0.f);
}

// ---------------- max pool over nsample ----------------
__global__ void kmaxpool(const float* __restrict__ Y, int O, int P, int S, int Lg,
                         float* __restrict__ feat) {
  int i = blockIdx.x * blockDim.x + threadIdx.x;
  if (i >= O * P) return;
  int o = i / P, p = i % P;
  const float* row = Y + (size_t)o * Lg + (size_t)p * S;
  float m = row[0];
  for (int s = 1; s < S; s++) m = fmaxf(m, row[s]);
  feat[(size_t)p * O + o] = m;
}

// ---------------- FP: 3-NN + inverse-distance weights ----------------
__global__ void kfp3nn(const float* __restrict__ dxyz, int Nd,
                       const float* __restrict__ sxyz, int Ms,
                       int* __restrict__ idx3, float* __restrict__ w3) {
  int i = blockIdx.x * blockDim.x + threadIdx.x;
  if (i >= B_SZ * Nd) return;
  int b = i / Nd;
  float qx = dxyz[i*3], qy = dxyz[i*3+1], qz = dxyz[i*3+2];
  const float* X = sxyz + (size_t)b * Ms * 3;
  float d0 = 1e30f, d1 = 1e30f, d2 = 1e30f; int i0 = 0, i1 = 0, i2 = 0;
  for (int n = 0; n < Ms; n++) {
    float dx = X[n*3]-qx, dy = X[n*3+1]-qy, dz = X[n*3+2]-qz;
    float d = dx*dx + dy*dy + dz*dz;
    if (d < d0)      { d2=d1; i2=i1; d1=d0; i1=i0; d0=d; i0=n; }
    else if (d < d1) { d2=d1; i2=i1; d1=d;  i1=n; }
    else if (d < d2) { d2=d;  i2=n; }
  }
  float w0 = 1.f/(fmaxf(d0,0.f)+1e-8f);
  float w1 = 1.f/(fmaxf(d1,0.f)+1e-8f);
  float w2 = 1.f/(fmaxf(d2,0.f)+1e-8f);
  float s = w0 + w1 + w2;
  idx3[i*3] = i0; idx3[i*3+1] = i1; idx3[i*3+2] = i2;
  w3[i*3] = w0/s; w3[i*3+1] = w1/s; w3[i*3+2] = w2/s;
}

// ---------------- FP: build bf16 X = [interp(featS) ; featD ; 0pad] ----------------
__global__ void kfp_build(const int* __restrict__ idx3, const float* __restrict__ w3,
                          const float* __restrict__ featS, int Cs, int Ms,
                          const float* __restrict__ featD, int Cd, int Nd,
                          uint16_t* __restrict__ Xbf, int Cpad) {
  int i = blockIdx.x * blockDim.x + threadIdx.x;
  if (i >= B_SZ * Nd) return;
  int b = i / Nd;
  int j0 = idx3[i*3], j1 = idx3[i*3+1], j2 = idx3[i*3+2];
  float w0 = w3[i*3], w1 = w3[i*3+1], w2 = w3[i*3+2];
  const float* f0 = featS + ((size_t)b*Ms + j0) * Cs;
  const float* f1 = featS + ((size_t)b*Ms + j1) * Cs;
  const float* f2 = featS + ((size_t)b*Ms + j2) * Cs;
  uint16_t* o = Xbf + (size_t)i * Cpad;
  for (int c = 0; c < Cs; c++) o[c] = f2bf(w0*f0[c] + w1*f1[c] + w2*f2[c]);
  const float* fd = featD + (size_t)i * Cd;
  for (int c = 0; c < Cd; c++)        o[Cs+c] = f2bf(fd[c]);
  for (int c = Cs+Cd; c < Cpad; c++)  o[c] = 0;
}

// ---------------- f32 point-major feat -> bf16 X ----------------
__global__ void kfeat_to_xbf(const float* __restrict__ feat, int C, int Ltot,
                             uint16_t* __restrict__ Xbf, int Cpad) {
  size_t i = (size_t)blockIdx.x * blockDim.x + threadIdx.x;
  if (i >= (size_t)Ltot * Cpad) return;
  int l = (int)(i / Cpad), c = (int)(i % Cpad);
  Xbf[i] = f2bf(c < C ? feat[(size_t)l * C + c] : 0.f);
}

// ---------------- final: Y[32][B*N] rows 0..12 -> out (B,13,N) ----------------
__global__ void kfc_out(const float* __restrict__ Y, float* __restrict__ out) {
  int i = blockIdx.x * blockDim.x + threadIdx.x;
  if (i >= B_SZ * 13 * N0) return;
  int n = i % N0; int c = (i / N0) % 13; int b = i / (13 * N0);
  out[i] = Y[(size_t)c * (B_SZ * N0) + (size_t)b * N0 + n];
}

// ============================================================================
extern "C" void kernel_launch(void* const* d_in, const int* in_sizes, int n_in,
                              void* d_out, int out_size, void* d_ws, size_t ws_size,
                              hipStream_t stream) {
  (void)in_sizes; (void)n_in; (void)out_size; (void)ws_size;
  const float* pc = (const float*)d_in[0];

  static const int SA_D[4][4] = {{6,32,32,64},{67,64,64,128},{131,128,128,256},{259,256,256,512}};
  static const int FP_D[4][4] = {{131,128,128,128},{320,256,128,0},{384,256,256,0},{768,256,256,0}};
  static const int FP_NC[4]   = {3,2,2,2};
  static const int Npts[5]    = {4096,1024,256,64,16};
  static const int FeatC[5]   = {3,64,128,256,512};
  static const int UC[4]      = {128,128,256,256};   // FP output channels per level

  // --- parameter pointers in setup_inputs() dict order: pointcloud, sa, fp, fc ---
  int pi = 1;
  const float *saW[4][3], *saB[4][3];
  for (int i = 0; i < 4; i++)
    for (int j = 0; j < 3; j++) { saW[i][j] = (const float*)d_in[pi++]; saB[i][j] = (const float*)d_in[pi++]; }
  const float *fpW[4][3] = {}, *fpB[4][3] = {};
  for (int i = 0; i < 4; i++)
    for (int j = 0; j < FP_NC[i]; j++) { fpW[i][j] = (const float*)d_in[pi++]; fpB[i][j] = (const float*)d_in[pi++]; }
  const float *fcW[2], *fcB[2];
  for (int j = 0; j < 2; j++) { fcW[j] = (const float*)d_in[pi++]; fcB[j] = (const float*)d_in[pi++]; }

  // --- workspace bump allocator (~250 MB total) ---
  char* base = (char*)d_ws; size_t off = 0;
  auto alloc = [&](size_t n) -> void* { void* p = base + off; off = (off + n + 255) & ~(size_t)255; return p; };
  float* xyzL[5]; for (int i = 0; i < 5; i++) xyzL[i] = (float*)alloc((size_t)B_SZ * Npts[i] * 3 * 4);
  float* featL[5]; for (int i = 0; i < 5; i++) featL[i] = (float*)alloc((size_t)B_SZ * Npts[i] * FeatC[i] * 4);
  float* featU[4]; for (int i = 0; i < 4; i++) featU[i] = (float*)alloc((size_t)B_SZ * Npts[i] * UC[i] * 4);
  int*   fpsidx = (int*)alloc((size_t)B_SZ * 1024 * 4);
  int*   nidx   = (int*)alloc((size_t)B_SZ * 1024 * 32 * 4);
  int*   idx3   = (int*)alloc((size_t)B_SZ * 4096 * 3 * 4);
  float* w3     = (float*)alloc((size_t)B_SZ * 4096 * 3 * 4);
  float* meanB  = (float*)alloc(512 * 4);
  float* rstdB  = (float*)alloc(512 * 4);
  uint16_t* WbfB = (uint16_t*)alloc((size_t)512 * 768 * 2);
  uint16_t* Xb[2];
  Xb[0] = (uint16_t*)alloc((size_t)16777216 * 2);   // max Lg*Cpad = 524288*32
  Xb[1] = (uint16_t*)alloc((size_t)16777216 * 2);
  float* Y = (float*)alloc((size_t)33554432 * 4);   // max Opad*Lg = 64*524288

  auto conv = [&](const float* Wp, const float* bp, int O, int C, const uint16_t* Xin, int L) {
    int Opad = (O + 31) & ~31, Cp = (C + 31) & ~31;   // O padded to 32 (wave computes 32 rows)
    int nW = Opad * Cp;
    kconvw<<<(nW + 255) / 256, 256, 0, stream>>>(Wp, O, C, WbfB, Opad, Cp);
    int nO2 = Opad / 32, nL2 = L / 32, lblocks = (nL2 + 7) / 8;
    kgemm<<<nO2 * lblocks, 256, 0, stream>>>(WbfB, Xin, bp, O, Y, Opad, Cp, L);
    kbnstats<<<O, 256, 0, stream>>>(Y, L, meanB, rstdB);
  };

  // ---- level 0 ----
  kextract<<<(B_SZ * N0 + 255) / 256, 256, 0, stream>>>(pc, xyzL[0], featL[0], B_SZ * N0);

  // ---- SA stages ----
  for (int li = 0; li < 4; li++) {
    int Nin = Npts[li], M = Npts[li + 1], Cin = FeatC[li];
    kfps<<<B_SZ, 256, 0, stream>>>(xyzL[li], Nin, M, fpsidx);
    kgather_xyz<<<(B_SZ * M + 255) / 256, 256, 0, stream>>>(xyzL[li], Nin, fpsidx, M, xyzL[li + 1]);
    kknn<<<(B_SZ * M + 255) / 256, 256, 0, stream>>>(xyzL[li + 1], M, xyzL[li], Nin, nidx);
    int Lg = B_SZ * M * 32;
    int Cp0 = (SA_D[li][0] + 31) & ~31;
    kgroup<<<(Lg + 255) / 256, 256, 0, stream>>>(xyzL[li], Nin, xyzL[li + 1], M, nidx,
                                                 featL[li], Cin, Xb[0], Cp0);
    int cur = 0;
    for (int j = 0; j < 3; j++) {
      int Ci = SA_D[li][j], O = SA_D[li][j + 1];
      conv(saW[li][j], saB[li][j], O, Ci, Xb[cur], Lg);
      if (j < 2) {
        int Cp2 = (O + 31) & ~31;
        size_t tot = (size_t)Cp2 * Lg;
        kbnrelu_xbf<<<(unsigned)((tot + 255) / 256), 256, 0, stream>>>(Y, O, Lg, meanB, rstdB, Xb[1 - cur], Cp2);
        cur = 1 - cur;
      } else {
        size_t tot = (size_t)O * Lg;
        kbnrelu_y<<<(unsigned)((tot + 255) / 256), 256, 0, stream>>>(Y, O, Lg, meanB, rstdB);
        int P = B_SZ * M;
        kmaxpool<<<(O * P + 255) / 256, 256, 0, stream>>>(Y, O, P, 32, Lg, featL[li + 1]);
      }
    }
  }

  // ---- FP stages (coarse -> fine): d = 3,2,1,0 uses params['fp'][d] ----
  for (int t = 0; t < 4; t++) {
    int d = 3 - t;
    int Nd = Npts[d], Ms = Npts[d + 1];
    const float* fS = (d == 3) ? featL[4] : featU[d + 1];
    int Cs = (d == 3) ? 512 : UC[d + 1];
    int Cd = FeatC[d];
    kfp3nn<<<(B_SZ * Nd + 255) / 256, 256, 0, stream>>>(xyzL[d], Nd, xyzL[d + 1], Ms, idx3, w3);
    int Cp = (FP_D[d][0] + 31) & ~31;
    int L = B_SZ * Nd;
    kfp_build<<<(B_SZ * Nd + 255) / 256, 256, 0, stream>>>(idx3, w3, fS, Cs, Ms,
                                                           featL[d], Cd, Nd, Xb[0], Cp);
    int cur = 0, nc = FP_NC[d];
    for (int j = 0; j < nc; j++) {
      int Ci = FP_D[d][j], O = FP_D[d][j + 1];
      conv(fpW[d][j], fpB[d][j], O, Ci, Xb[cur], L);
      if (j < nc - 1) {
        int Cp2 = (O + 31) & ~31;
        size_t tot = (size_t)Cp2 * L;
        kbnrelu_xbf<<<(unsigned)((tot + 255) / 256), 256, 0, stream>>>(Y, O, L, meanB, rstdB, Xb[1 - cur], Cp2);
        cur = 1 - cur;
      } else {
        size_t tot = (size_t)O * L;
        kbnrelu_feat<<<(unsigned)((tot + 255) / 256), 256, 0, stream>>>(Y, O, L, meanB, rstdB, featU[d]);
      }
    }
  }

  // ---- FC head ----
  int L0 = B_SZ * N0;
  kfeat_to_xbf<<<(unsigned)(((size_t)L0 * 128 + 255) / 256), 256, 0, stream>>>(featU[0], 128, L0, Xb[0], 128);
  conv(fcW[0], fcB[0], 128, 128, Xb[0], L0);
  {
    size_t tot = (size_t)128 * L0;
    kbnrelu_xbf<<<(unsigned)((tot + 255) / 256), 256, 0, stream>>>(Y, 128, L0, meanB, rstdB, Xb[1], 128);
  }
  conv(fcW[1], fcB[1], 13, 128, Xb[1], L0);   // bias added in GEMM epilogue; no BN on final layer
  kfc_out<<<(B_SZ * 13 * N0 + 255) / 256, 256, 0, stream>>>(Y, (float*)d_out);
}
